// TEmbResBlock_77283641524428
// MI455X (gfx1250) — matrix operand order repaired and checked by reading, not compiled, not run on hardware
//
#include <hip/hip_runtime.h>
#include <hip/hip_bf16.h>
#include <stddef.h>
#include <stdint.h>

// ---------------------------------------------------------------------------
// Problem constants (from reference)
// ---------------------------------------------------------------------------
#define B_   32
#define C_   128
#define H_   64
#define W_   64
#define HW_  (H_ * W_)          // 4096
#define NPIX (B_ * HW_)         // 131072
#define NELT (B_ * C_ * HW_)    // 16,777,216
#define DIM_ 512
#define EPS_ 1e-5f

// ---------------------------------------------------------------------------
// Vector types for WMMA
// ---------------------------------------------------------------------------
typedef __attribute__((ext_vector_type(8)))  float          v8f;
typedef __attribute__((ext_vector_type(16))) __bf16         v16bf;
typedef __attribute__((ext_vector_type(8)))  __bf16         v8bf;
typedef __attribute__((ext_vector_type(8)))  unsigned short u16x8;

union ABFrag {
    v16bf v;
    u16x8 u[2];
};

__device__ __forceinline__ unsigned short f32_to_bf16(float f) {
    unsigned int u = __float_as_uint(f);
    unsigned int r = u + 0x7FFFu + ((u >> 16) & 1u); // round-to-nearest-even
    return (unsigned short)(r >> 16);
}

// ---------------------------------------------------------------------------
// 1) Time embedding: te[b][c] = silu( sum_i emb(T[b], i) * lin_w[c][i] + lin_b[c] )
// ---------------------------------------------------------------------------
__global__ void temb_kernel(const float* __restrict__ T,
                            const float* __restrict__ lin_w,
                            const float* __restrict__ lin_b,
                            float* __restrict__ te) {
    const int b = blockIdx.x;   // 32
    const int c = threadIdx.x;  // 128
    const float t = T[b];
    float acc = 0.f;
    const float* wrow = lin_w + (size_t)c * DIM_;
#pragma unroll 4
    for (int i = 0; i < DIM_; ++i) {
        float ang = t * expf(-logf(10000.0f) * (2.0f * (float)i) / (float)DIM_);
        float e = (i & 1) ? cosf(ang) : sinf(ang);
        acc += e * wrow[i];
    }
    acc += lin_b[c];
    te[b * C_ + c] = acc / (1.0f + expf(-acc)); // SiLU
}

// ---------------------------------------------------------------------------
// 2) Pack conv weights fp32 OIHW -> bf16 [blk][kk=kh*3+kw][co][ci]
// ---------------------------------------------------------------------------
__global__ void pack_w_kernel(const float* __restrict__ cw,
                              unsigned short* __restrict__ wp) {
    int idx = blockIdx.x * 256 + threadIdx.x;      // 2*9*128*128 = 294912
    if (idx >= 2 * 9 * C_ * C_) return;
    int ci  = idx & 127;
    int co  = (idx >> 7) & 127;
    int kk  = (idx >> 14) % 9;
    int blk = (idx >> 14) / 9;
    int kh = kk / 3, kw = kk % 3;
    float v = cw[((((size_t)blk * C_ + co) * C_ + ci) * 3 + kh) * 3 + kw];
    wp[idx] = f32_to_bf16(v);
}

// ---------------------------------------------------------------------------
// 3) NCHW fp32 -> NHWC bf16
// ---------------------------------------------------------------------------
__global__ void cvt_x_kernel(const float* __restrict__ x,
                             unsigned short* __restrict__ xb) {
    size_t idx = (size_t)blockIdx.x * 256 + threadIdx.x; // NHWC index
    int c  = (int)(idx & 127);
    size_t p = idx >> 7;          // b*HW + hw
    size_t hw = p & (HW_ - 1);
    size_t b  = p >> 12;
    xb[idx] = f32_to_bf16(x[((b * C_ + c) << 12) + hw]);
}

// ---------------------------------------------------------------------------
// 4) Implicit-GEMM conv 3x3 via WMMA bf16 -> fp32, y = conv(x) + bias (NCHW fp32)
//    Tile: 128 co x 128 pixels (2 output rows), 8 waves.
//    Input slab staged with GLOBAL_LOAD_ASYNC_TO_LDS_B128 (ASYNCcnt path);
//    zero halo written with DS stores.
// ---------------------------------------------------------------------------
__global__ __launch_bounds__(256) void conv_gemm_kernel(
        const unsigned short* __restrict__ xb,  // NHWC bf16 input
        const unsigned short* __restrict__ wp,  // [9][128][128] bf16 (one block)
        const float* __restrict__ bias,         // [128]
        float* __restrict__ y) {                // NCHW fp32 pre-BN out
    // LDS slab: 4 input rows (h0-1..h0+2) x 66 cols (-1..64) x 64 ci, bf16
    __shared__ unsigned short slab[4 * 66 * 64];

    const int tid    = threadIdx.x;
    const int lane   = tid & 31;
    const int wave   = tid >> 5;          // 0..7 -> co stripe
    const int mrow   = lane & 15;
    const int hiHalf = lane >> 4;         // 0: lanes 0-15, 1: lanes 16-31

    const int bidx = blockIdx.x;          // B_*(H_/2) = 1024
    const int b    = bidx >> 5;
    const int h0   = (bidx & 31) << 1;    // output rows h0, h0+1

    v8f acc[8];
#pragma unroll
    for (int t = 0; t < 8; ++t) {
        v8f z;
#pragma unroll
        for (int r = 0; r < 8; ++r) z[r] = 0.0f;
        acc[t] = z;
    }

    for (int cchunk = 0; cchunk < C_; cchunk += 64) {
        __syncthreads(); // previous chunk readers done before re-staging
        // ---- stage zero-padded input slab: 4*66*8 = 2112 16-byte transfers ----
        for (int v = tid; v < 4 * 66 * 8; v += 256) {
            int cig = v & 7;              // group of 8 ci
            int cc  = (v >> 3) % 66;      // slab col (input col cc-1)
            int rr  = (v >> 3) / 66;      // slab row (input row h0-1+rr)
            int hs  = h0 - 1 + rr;
            int ws  = cc - 1;
            unsigned short* sp = slab + ((rr * 66 + cc) * 64 + cig * 8);
            if (hs >= 0 && hs < H_ && ws >= 0 && ws < W_) {
                // async direct-to-LDS copy, tracked on ASYNCcnt (no VGPR round-trip)
                unsigned int lds_off = (unsigned int)(uintptr_t)sp;
                unsigned long long ga = (unsigned long long)(uintptr_t)(xb +
                        ((((size_t)b * H_ + hs) * W_ + ws) * C_ + cchunk + cig * 8));
                asm volatile("global_load_async_to_lds_b128 %0, %1, off"
                             :: "v"(lds_off), "v"(ga) : "memory");
            } else {
                u16x8 z = {0, 0, 0, 0, 0, 0, 0, 0};
                *(u16x8*)sp = z;          // zero halo via DS store (DScnt)
            }
        }
        asm volatile("s_wait_asynccnt 0x0" ::: "memory");
        __syncthreads();

        // ---- K loop: 9 taps x 2 wmma steps over this 64-ci chunk ----
        for (int kk = 0; kk < 9; ++kk) {
            const int kh = kk / 3, kw = kk % 3;
            // prefetch next tap's weight stripe into L2
            if (kk < 8) {
                __builtin_prefetch(wp + (((size_t)(kk + 1) * C_ + wave * 16 + mrow) * C_
                                         + cchunk), 0, 1);
            }
#pragma unroll
            for (int kci = 0; kci < 64; kci += 32) {
                // A fragment: 16 co x 32 ci of weights (ISA 16-bit A layout)
                const int co  = wave * 16 + mrow;
                const int ci0 = cchunk + kci + (hiHalf ? 8 : 0);
                const unsigned short* wa = wp + (((size_t)kk * C_ + co) * C_ + ci0);
                ABFrag a;
                a.u[0] = *(const u16x8*)(wa);       // k = {0..7}   (+8 if hi)
                a.u[1] = *(const u16x8*)(wa + 16);  // k = {16..23} (+8 if hi)

                const int kb = kci + (hiHalf ? 8 : 0); // B lane k-base (table order)
#pragma unroll
                for (int t = 0; t < 8; ++t) {
                    // B fragment: 32 ci x 16 pixels from LDS slab
                    int n  = t * 16 + mrow;             // pixel-in-tile (column n)
                    int rr = (n >> 6) + kh;             // shifted slab row
                    int cc = (n & 63) + kw;             // shifted slab col
                    const unsigned short* bp = slab + ((rr * 66 + cc) * 64 + kb);
                    ABFrag bf;
                    bf.u[0] = *(const u16x8*)(bp);      // k = kb .. kb+7
                    bf.u[1] = *(const u16x8*)(bp + 16); // k = kb+16 .. kb+23
                    acc[t] = __builtin_amdgcn_wmma_f32_16x16x32_bf16(
                        false, a.v, false, bf.v, (short)0, acc[t], false, false);
                }
            }
        }
    }

    // ---- writeback: D layout (16x16 f32, 8 VGPRs) -> NCHW fp32 + bias ----
#pragma unroll
    for (int t = 0; t < 8; ++t) {
        int n    = t * 16 + mrow;
        int h    = h0 + (n >> 6);
        int wcol = n & 63;
#pragma unroll
        for (int r = 0; r < 8; ++r) {
            int co = wave * 16 + r + (hiHalf ? 8 : 0);
            y[(((size_t)b * C_ + co) << 12) + h * W_ + wcol] = acc[t][r] + bias[co];
        }
    }
}

// ---------------------------------------------------------------------------
// 5a) BN partial sums: 8 partial blocks per channel (deterministic, no atomics)
//     part[c*8+p] = sum over b in [4p,4p+4), all hw; part[1024+..] = sumsq
// ---------------------------------------------------------------------------
__global__ __launch_bounds__(256) void bn_partial_kernel(const float* __restrict__ y,
                                                         float* __restrict__ part) {
    const int c  = blockIdx.x >> 3;   // channel
    const int pb = blockIdx.x & 7;    // partial index
    const int tid = threadIdx.x;
    float s = 0.f, sq = 0.f;
    for (int i = tid; i < 4 * HW_; i += 256) {
        int b  = pb * 4 + (i >> 12);
        int hw = i & (HW_ - 1);
        float v = y[(((size_t)b * C_ + c) << 12) + hw];
        s  += v;
        sq += v * v;
    }
    __shared__ float sh[512];
    sh[tid] = s;
    sh[tid + 256] = sq;
    __syncthreads();
    for (int off = 128; off > 0; off >>= 1) {
        if (tid < off) {
            sh[tid]       += sh[tid + off];
            sh[tid + 256] += sh[tid + 256 + off];
        }
        __syncthreads();
    }
    if (tid == 0) {
        part[blockIdx.x]        = sh[0];
        part[1024 + blockIdx.x] = sh[256];
    }
}

// ---------------------------------------------------------------------------
// 5b) BN finalize: stats[c]=mean, stats[128+c]=rstd
// ---------------------------------------------------------------------------
__global__ void bn_finalize_kernel(const float* __restrict__ part,
                                   float* __restrict__ stats) {
    const int c = threadIdx.x; // 128
    float s = 0.f, sq = 0.f;
#pragma unroll
    for (int p = 0; p < 8; ++p) {
        s  += part[c * 8 + p];
        sq += part[1024 + c * 8 + p];
    }
    const float inv_n = 1.0f / (float)NPIX;
    float mean = s * inv_n;
    float var  = sq * inv_n - mean * mean;
    stats[c]      = mean;
    stats[C_ + c] = rsqrtf(var + EPS_);
}

// ---------------------------------------------------------------------------
// 6a) BN apply (mid): h = relu(bn(y)) + te  -> bf16 NHWC (input of block 1)
// ---------------------------------------------------------------------------
__global__ void bn_apply_mid_kernel(const float* __restrict__ y,
                                    const float* __restrict__ stats,
                                    const float* __restrict__ g,
                                    const float* __restrict__ be,
                                    const float* __restrict__ te,
                                    unsigned short* __restrict__ hb) {
    size_t idx = (size_t)blockIdx.x * 256 + threadIdx.x; // NCHW index
    int hw = (int)(idx & (HW_ - 1));
    int c  = (int)((idx >> 12) & 127);
    int b  = (int)(idx >> 19);
    float v = y[idx];
    v = (v - stats[c]) * stats[C_ + c] * g[c] + be[c];
    v = fmaxf(v, 0.f) + te[b * C_ + c];
    hb[(((size_t)b << 12) + hw) * C_ + c] = f32_to_bf16(v);
}

// ---------------------------------------------------------------------------
// 6b) BN apply (final): out = relu(bn(y)) + te + x   (fp32 NCHW)
// ---------------------------------------------------------------------------
__global__ void bn_apply_final_kernel(const float* __restrict__ y,
                                      const float* __restrict__ stats,
                                      const float* __restrict__ g,
                                      const float* __restrict__ be,
                                      const float* __restrict__ te,
                                      const float* __restrict__ x,
                                      float* __restrict__ out) {
    size_t idx = (size_t)blockIdx.x * 256 + threadIdx.x; // NCHW index
    int c = (int)((idx >> 12) & 127);
    int b = (int)(idx >> 19);
    float v = y[idx];
    v = (v - stats[c]) * stats[C_ + c] * g[c] + be[c];
    v = fmaxf(v, 0.f) + te[b * C_ + c];
    out[idx] = v + x[idx];
}

// ---------------------------------------------------------------------------
// Launch: temb | pack_w | cvt | [conv -> stats(2-stage) -> apply] x2
// ---------------------------------------------------------------------------
extern "C" void kernel_launch(void* const* d_in, const int* in_sizes, int n_in,
                              void* d_out, int out_size, void* d_ws, size_t ws_size,
                              hipStream_t stream) {
    const float* x      = (const float*)d_in[0];
    const float* T      = (const float*)d_in[1];
    const float* conv_w = (const float*)d_in[2];
    const float* conv_b = (const float*)d_in[3];
    const float* gamma  = (const float*)d_in[4];
    const float* beta   = (const float*)d_in[5];
    const float* lin_w  = (const float*)d_in[6];
    const float* lin_b  = (const float*)d_in[7];
    float* out          = (float*)d_out;

    // Workspace layout
    char* ws = (char*)d_ws;
    float*          y     = (float*)ws;                                   // 64 MiB
    unsigned short* xb    = (unsigned short*)(ws + (size_t)NELT * 4);     // 32 MiB (also hb)
    unsigned short* wp    = (unsigned short*)(ws + (size_t)NELT * 4 + (size_t)NELT * 2);
    float*          te    = (float*)((char*)wp + (size_t)2 * 9 * C_ * C_ * 2);
    float*          stats = (float*)((char*)te + (size_t)B_ * C_ * 4);
    float*          part  = (float*)((char*)stats + 2 * C_ * 4);          // 2048 floats

    const int blkElems = NELT / 256;       // 65536 blocks for elementwise passes

    temb_kernel<<<B_, C_, 0, stream>>>(T, lin_w, lin_b, te);
    pack_w_kernel<<<(2 * 9 * C_ * C_ + 255) / 256, 256, 0, stream>>>(conv_w, wp);
    cvt_x_kernel<<<blkElems, 256, 0, stream>>>(x, xb);

    // ---- Block 0 ----
    conv_gemm_kernel<<<B_ * (H_ / 2), 256, 0, stream>>>(xb, wp, conv_b, y);
    bn_partial_kernel<<<C_ * 8, 256, 0, stream>>>(y, part);
    bn_finalize_kernel<<<1, C_, 0, stream>>>(part, stats);
    bn_apply_mid_kernel<<<blkElems, 256, 0, stream>>>(y, stats, gamma, beta, te, xb);

    // ---- Block 1 (input = xb, now holding h0 in NHWC bf16) ----
    conv_gemm_kernel<<<B_ * (H_ / 2), 256, 0, stream>>>(xb, wp + (size_t)9 * C_ * C_,
                                                        conv_b + C_, y);
    bn_partial_kernel<<<C_ * 8, 256, 0, stream>>>(y, part);
    bn_finalize_kernel<<<1, C_, 0, stream>>>(part, stats);
    bn_apply_final_kernel<<<blkElems, 256, 0, stream>>>(y, stats, gamma + C_, beta + C_,
                                                        te, x, out);
    (void)in_sizes; (void)n_in; (void)out_size; (void)ws_size;
}
